// QwenAttention_23957327577831
// MI455X (gfx1250) — matrix operand order repaired and checked by reading, not compiled, not run on hardware
//
#include <hip/hip_runtime.h>

#define HIDW  2048
#define NH    16
#define HD    128
#define SEQ   2048
#define BATCH 2
#define ROWS  (BATCH*SEQ)   // 4096

typedef _Float16 f16;
typedef _Float16 v16h __attribute__((ext_vector_type(16)));
typedef _Float16 v8h  __attribute__((ext_vector_type(8)));
typedef float    v8f  __attribute__((ext_vector_type(8)));

__device__ __forceinline__ v16h cat8(v8h lo, v8h hi) {
    return __builtin_shufflevector(lo, hi, 0,1,2,3,4,5,6,7,8,9,10,11,12,13,14,15);
}
__device__ __forceinline__ v8f wmma_f16(v16h a, v16h b, v8f c) {
    return __builtin_amdgcn_wmma_f32_16x16x32_f16(false, a, false, b, (short)0, c, false, false);
}
// Async global->LDS copy (ASYNCcnt-tracked), 16 bytes per lane.
__device__ __forceinline__ void async_copy_b128(const f16* gsrc, f16* lds_dst) {
    unsigned lds_addr = (unsigned)(unsigned long long)(void*)lds_dst; // LDS offset = addr[31:0]
    unsigned long long gaddr = (unsigned long long)(const void*)gsrc;
    asm volatile("global_load_async_to_lds_b128 %0, %1, off"
                 :: "v"(lds_addr), "v"(gaddr) : "memory");
}
__device__ __forceinline__ void wait_async() {
    asm volatile("s_wait_asynccnt 0" ::: "memory");
}

// ---------------- elementwise f32 -> f16 ----------------
struct H4 { f16 a, b, c, d; };
__global__ void cvt_f32_to_f16(const float* __restrict__ x, f16* __restrict__ y, int n4) {
    int i = blockIdx.x * blockDim.x + threadIdx.x;
    if (i >= n4) return;
    float4 v = ((const float4*)x)[i];
    H4 o; o.a = (f16)v.x; o.b = (f16)v.y; o.c = (f16)v.z; o.d = (f16)v.w;
    ((H4*)y)[i] = o;
}

// ---------------- transpose + convert: W[k][n] f32 -> WT[n][k] f16 ----------------
__global__ void cvt_transpose(const float* __restrict__ w, f16* __restrict__ wt) {
    __shared__ float tile[32][33];
    const int k0 = blockIdx.y * 32, n0 = blockIdx.x * 32;
    const int tx = threadIdx.x, ty = threadIdx.y;          // 32 x 8
    for (int r = 0; r < 32; r += 8)
        tile[ty + r][tx] = w[(size_t)(k0 + ty + r) * HIDW + n0 + tx];
    __syncthreads();
    for (int r = 0; r < 32; r += 8)
        wt[(size_t)(n0 + ty + r) * HIDW + k0 + tx] = (f16)tile[tx][ty + r];
}

// ---------------- WMMA GEMM: Y = A(ROWSxK) * WT^T (+bias) ----------------
// Block tile 64x128x32, 8 waves in 2x4; double-buffered async LDS pipeline.
// mode 0: f16 out, [b][h][s][d]; mode 1: f16 out, [b][h][d][s]; mode 2: f32 + bias.
__global__ void __launch_bounds__(256)
gemm_wmma(const f16* __restrict__ A, const f16* __restrict__ Wt,
          const float* __restrict__ bias,
          f16* __restrict__ out16, float* __restrict__ out32, int mode) {
    __shared__ __align__(16) f16 As[2][64][40];
    __shared__ __align__(16) f16 Bs[2][128][40];
    const int tid  = threadIdx.x;
    const int wave = tid >> 5;
    const int lane = tid & 31;
    const int m0 = blockIdx.y * 64, n0 = blockIdx.x * 128;
    const int wm = (wave >> 2) * 32;
    const int wn = (wave & 3) * 32;
    const int col16 = lane & 15, hsel = lane >> 4;
    const int ar = tid >> 2, ac = (tid & 3) * 8;    // A tile: 64 x 32 halves
    const int br = tid >> 1, bc = (tid & 1) * 16;   // B tile: 128 x 32 halves
    const int NSTEP = HIDW / 32;
    v8f acc[2][2];
    acc[0][0] = {}; acc[0][1] = {}; acc[1][0] = {}; acc[1][1] = {};

    // prologue: stage k-step 0 into buffer 0
    async_copy_b128(&A [(size_t)(m0 + ar) * HIDW + ac], &As[0][ar][ac]);
    async_copy_b128(&Wt[(size_t)(n0 + br) * HIDW + bc],     &Bs[0][br][bc]);
    async_copy_b128(&Wt[(size_t)(n0 + br) * HIDW + bc + 8], &Bs[0][br][bc + 8]);

    for (int it = 0; it < NSTEP; ++it) {
        const int cur = it & 1;
        wait_async();        // this wave's stage-it copies done
        __syncthreads();     // all copies visible; all prior-stage reads retired
        if (it + 1 < NSTEP) {
            const int kn = (it + 1) * 32, nxt = cur ^ 1;
            async_copy_b128(&A [(size_t)(m0 + ar) * HIDW + kn + ac], &As[nxt][ar][ac]);
            async_copy_b128(&Wt[(size_t)(n0 + br) * HIDW + kn + bc],     &Bs[nxt][br][bc]);
            async_copy_b128(&Wt[(size_t)(n0 + br) * HIDW + kn + bc + 8], &Bs[nxt][br][bc + 8]);
        }
        v16h af[2], bf[2];
        for (int mi = 0; mi < 2; ++mi) {
            const f16* p = &As[cur][wm + mi * 16 + col16][hsel * 8];
            af[mi] = cat8(*(const v8h*)p, *(const v8h*)(p + 16));
        }
        for (int ni = 0; ni < 2; ++ni) {
            const f16* p = &Bs[cur][wn + ni * 16 + col16][hsel * 8];
            bf[ni] = cat8(*(const v8h*)p, *(const v8h*)(p + 16));
        }
        for (int mi = 0; mi < 2; ++mi)
            for (int ni = 0; ni < 2; ++ni)
                acc[mi][ni] = wmma_f16(af[mi], bf[ni], acc[mi][ni]);
    }
    for (int mi = 0; mi < 2; ++mi)
        for (int ni = 0; ni < 2; ++ni) {
            const int col = n0 + wn + ni * 16 + col16;
            const float bv = bias ? bias[col] : 0.f;
            for (int vg = 0; vg < 8; ++vg) {
                const int row = m0 + wm + mi * 16 + vg + 8 * hsel;
                float v = acc[mi][ni][vg] + bv;
                if (mode == 2) {
                    out32[(size_t)row * HIDW + col] = v;
                } else {
                    const int b = row >> 11, s = row & (SEQ - 1);
                    const int h = col >> 7,  d = col & (HD - 1);
                    size_t idx = (mode == 0)
                        ? ((((size_t)b * NH + h) * SEQ + s) * HD + d)
                        : ((((size_t)b * NH + h) * HD + d) * SEQ + s);
                    out16[idx] = (f16)v;
                }
            }
        }
}

// ---------------- RoPE in-place on Q and K ([b][h][s][d] f16) ----------------
__global__ void rope_kernel(f16* __restrict__ Q, f16* __restrict__ K, int total) {
    int idx = blockIdx.x * blockDim.x + threadIdx.x;
    if (idx >= total) return;
    const int i   = idx & 63;
    const int row = idx >> 6;
    const int s   = row & (SEQ - 1);
    const float inv = __expf(-(float)i * (9.210340371976184f / 64.f)); // 10000^(-i/64)
    const float ang = (float)s * inv;
    const float c = cosf(ang), sn = sinf(ang);
    const size_t base = (size_t)row * HD;
    float q1 = (float)Q[base + i], q2 = (float)Q[base + 64 + i];
    Q[base + i]      = (f16)(q1 * c - q2 * sn);
    Q[base + 64 + i] = (f16)(q2 * c + q1 * sn);
    float k1 = (float)K[base + i], k2 = (float)K[base + 64 + i];
    K[base + i]      = (f16)(k1 * c - k2 * sn);
    K[base + 64 + i] = (f16)(k2 * c + k1 * sn);
}

// ---------------- flash attention ----------------
// 4 waves x 16 queries per block; K/V chunks staged in LDS via async double
// buffering and shared by all waves (4x less K/V read traffic).
__global__ void __launch_bounds__(128)
attn_kernel(const f16* __restrict__ Q, const f16* __restrict__ K,
            const f16* __restrict__ Vt, f16* __restrict__ ctx) {
    __shared__ __align__(16) f16 Ks[2][32][136];   // [key][d], padded pitch
    __shared__ __align__(16) f16 Vs[2][128][40];   // [d][key], padded pitch
    __shared__ __align__(16) f16 P [4][16][40];    // per-wave P tile
    const int tid = threadIdx.x;
    const int wave = tid >> 5, lane = tid & 31;
    const int blk = blockIdx.x, h = blockIdx.y, b = blockIdx.z;
    const int q0 = blk * 64 + wave * 16;
    const int col16 = lane & 15, hsel = lane >> 4;
    const size_t bh = (size_t)b * NH + h;
    const f16* Qb = Q  + bh * SEQ * HD;   // [s][d]
    const f16* Kb = K  + bh * SEQ * HD;   // [s][d]
    const f16* Vb = Vt + bh * HD * SEQ;   // [d][s]

    // cooperative staging indices (128 threads)
    const int kr = tid >> 2, kc = (tid & 3) * 32;  // Ks: 32 rows x 128 halves
    const int vr = tid;                            // Vs: 128 rows x 32 halves

    v16h qf[4];
    for (int c = 0; c < 4; ++c) {
        size_t base = (size_t)(q0 + col16) * HD + c * 32 + hsel * 8;
        qf[c] = cat8(*(const v8h*)&Qb[base], *(const v8h*)&Qb[base + 16]);
    }
    v8f acc[8];
    float mrun[8], lrun[8];
    for (int t = 0; t < 8; ++t) acc[t] = {};
    for (int vg = 0; vg < 8; ++vg) { mrun[vg] = -1e30f; lrun[vg] = 0.f; }
    const float scale = 0.08838834764831845f;   // 1/sqrt(128)
    const int nchunk = 2 * blk + 2;             // keys 0 .. blk*64+63 in 32-key chunks

    // prologue: stage chunk 0
    for (int off = 0; off < 32; off += 8)
        async_copy_b128(&Kb[(size_t)kr * HD + kc + off], &Ks[0][kr][kc + off]);
    for (int off = 0; off < 32; off += 8)
        async_copy_b128(&Vb[(size_t)vr * SEQ + off], &Vs[0][vr][off]);

    for (int it = 0; it < nchunk; ++it) {
        const int j0 = it * 32;
        const int cur = it & 1;
        wait_async();
        __syncthreads();
        if (it + 1 < nchunk) {
            const int jn = j0 + 32, nxt = cur ^ 1;
            for (int off = 0; off < 32; off += 8)
                async_copy_b128(&Kb[(size_t)(jn + kr) * HD + kc + off], &Ks[nxt][kr][kc + off]);
            for (int off = 0; off < 32; off += 8)
                async_copy_b128(&Vb[(size_t)vr * SEQ + jn + off], &Vs[nxt][vr][off]);
        }
        // scores: two 16x16 tiles over this 32-key chunk
        v8f sc[2];
        for (int nt = 0; nt < 2; ++nt) {
            v8f c = {};
            for (int cc = 0; cc < 4; ++cc) {
                const f16* p = &Ks[cur][nt * 16 + col16][cc * 32 + hsel * 8];
                v16h kf = cat8(*(const v8h*)p, *(const v8h*)(p + 16));
                c = wmma_f16(qf[cc], kf, c);
            }
            sc[nt] = c;
        }
        for (int vg = 0; vg < 8; ++vg) {
            const int qrow = q0 + vg + 8 * hsel;
            float s0 = sc[0][vg] * scale + ((j0 + col16)      > qrow ? -1e9f : 0.f);
            float s1 = sc[1][vg] * scale + ((j0 + 16 + col16) > qrow ? -1e9f : 0.f);
            float rm = fmaxf(s0, s1);
            for (int off = 1; off < 16; off <<= 1) rm = fmaxf(rm, __shfl_xor(rm, off, 32));
            float mn = fmaxf(mrun[vg], rm);
            float alpha = __expf(mrun[vg] - mn);
            float p0 = __expf(s0 - mn), p1 = __expf(s1 - mn);
            float rs = p0 + p1;
            for (int off = 1; off < 16; off <<= 1) rs += __shfl_xor(rs, off, 32);
            lrun[vg] = lrun[vg] * alpha + rs;
            mrun[vg] = mn;
            for (int t = 0; t < 8; ++t) acc[t][vg] *= alpha;
            const int prow = vg + 8 * hsel;
            P[wave][prow][col16]      = (f16)p0;
            P[wave][prow][16 + col16] = (f16)p1;
        }
        __syncthreads();   // uniform across waves; orders P write->read
        v16h pf = cat8(*(const v8h*)&P[wave][col16][hsel * 8],
                       *(const v8h*)&P[wave][col16][hsel * 8 + 16]);
        for (int t = 0; t < 8; ++t) {
            const f16* p = &Vs[cur][t * 16 + col16][hsel * 8];
            v16h vf = cat8(*(const v8h*)p, *(const v8h*)(p + 16));
            acc[t] = wmma_f16(pf, vf, acc[t]);
        }
    }
    for (int t = 0; t < 8; ++t)
        for (int vg = 0; vg < 8; ++vg) {
            const int qrow = q0 + vg + 8 * hsel;
            const int dcol = t * 16 + col16;
            ctx[((size_t)b * SEQ + qrow) * HIDW + h * HD + dcol] = (f16)(acc[t][vg] / lrun[vg]);
        }
}

extern "C" void kernel_launch(void* const* d_in, const int* in_sizes, int n_in,
                              void* d_out, int out_size, void* d_ws, size_t ws_size,
                              hipStream_t stream) {
    (void)in_sizes; (void)n_in; (void)out_size; (void)ws_size;
    const float* X  = (const float*)d_in[0];
    const float* Wq = (const float*)d_in[3];
    const float* bq = (const float*)d_in[4];
    const float* Wk = (const float*)d_in[5];
    const float* bk = (const float*)d_in[6];
    const float* Wv = (const float*)d_in[7];
    const float* bv = (const float*)d_in[8];
    const float* Wo = (const float*)d_in[9];
    const float* bo = (const float*)d_in[10];
    float* out = (float*)d_out;

    char* ws = (char*)d_ws;
    f16* Xh  = (f16*)ws; ws += (size_t)ROWS * HIDW * 2;
    f16* WqT = (f16*)ws; ws += (size_t)HIDW * HIDW * 2;
    f16* WkT = (f16*)ws; ws += (size_t)HIDW * HIDW * 2;
    f16* WvT = (f16*)ws; ws += (size_t)HIDW * HIDW * 2;
    f16* WoT = (f16*)ws; ws += (size_t)HIDW * HIDW * 2;
    f16* Qb  = (f16*)ws; ws += (size_t)ROWS * HIDW * 2;
    f16* Kb  = (f16*)ws; ws += (size_t)ROWS * HIDW * 2;
    f16* Vtb = (f16*)ws; ws += (size_t)ROWS * HIDW * 2;
    f16* Ctx = (f16*)ws; ws += (size_t)ROWS * HIDW * 2;

    const int n4 = ROWS * HIDW / 4;
    cvt_f32_to_f16<<<(n4 + 255) / 256, 256, 0, stream>>>(X, Xh, n4);
    dim3 tg(HIDW / 32, HIDW / 32), tb(32, 8);
    cvt_transpose<<<tg, tb, 0, stream>>>(Wq, WqT);
    cvt_transpose<<<tg, tb, 0, stream>>>(Wk, WkT);
    cvt_transpose<<<tg, tb, 0, stream>>>(Wv, WvT);
    cvt_transpose<<<tg, tb, 0, stream>>>(Wo, WoT);

    dim3 gg(HIDW / 128, ROWS / 64);
    gemm_wmma<<<gg, 256, 0, stream>>>(Xh, WqT, bq, Qb,  nullptr, 0);
    gemm_wmma<<<gg, 256, 0, stream>>>(Xh, WkT, bk, Kb,  nullptr, 0);
    gemm_wmma<<<gg, 256, 0, stream>>>(Xh, WvT, bv, Vtb, nullptr, 1);

    const int nr = BATCH * NH * SEQ * 64;
    rope_kernel<<<(nr + 255) / 256, 256, 0, stream>>>(Qb, Kb, nr);

    attn_kernel<<<dim3(SEQ / 64, NH, BATCH), 128, 0, stream>>>(Qb, Kb, Vtb, Ctx);

    gemm_wmma<<<gg, 256, 0, stream>>>(Ctx, WoT, bo, nullptr, out, 2);
}
